// Attention_11828339933424
// MI455X (gfx1250) — matrix operand order repaired and checked by reading, not compiled, not run on hardware
//
#include <hip/hip_runtime.h>
#include <hip/hip_bf16.h>

#define BATCH 4
#define SEQ 4096
#define IN_DIM 512
#define OUT_DIM 64

typedef __attribute__((ext_vector_type(16))) _Float16 v16h;
typedef __attribute__((ext_vector_type(8)))  _Float16 v8h;
typedef __attribute__((ext_vector_type(8)))  float    v8f;

// ---------------------------------------------------------------------------
// Kernel 0a: convert input embedding f32 -> f16
// ---------------------------------------------------------------------------
__global__ void k_cvt_x(const float* __restrict__ X, _Float16* __restrict__ Xh, int n) {
    for (int i = blockIdx.x * blockDim.x + threadIdx.x; i < n; i += gridDim.x * blockDim.x)
        Xh[i] = (_Float16)X[i];
}

// ---------------------------------------------------------------------------
// Kernel 0b: transpose+convert Wq/Wk/Wv [512,64] f32 -> Wth[m][64][512] f16
// ---------------------------------------------------------------------------
__global__ void k_wt(const float* __restrict__ Wq, const float* __restrict__ Wk,
                     const float* __restrict__ Wv, _Float16* __restrict__ Wth) {
    int id = blockIdx.x * blockDim.x + threadIdx.x;
    if (id >= 3 * OUT_DIM * IN_DIM) return;
    int m = id >> 15;          // /32768
    int r = id & 32767;
    int n = r >> 9;            // output col
    int k = r & 511;           // input dim
    const float* W = (m == 0) ? Wq : (m == 1) ? Wk : Wv;
    Wth[(size_t)m * 32768 + (size_t)n * 512 + k] = (_Float16)W[(size_t)k * 64 + n];
}

// ---------------------------------------------------------------------------
// Kernel 1: fused QKV projection (one wave -> 16x64 tile of Q,K,V; X read once)
// ---------------------------------------------------------------------------
__global__ void __launch_bounds__(128) k_qkv(
        const _Float16* __restrict__ Xh,   // [16384,512]
        const _Float16* __restrict__ Wth,  // [3][64][512]
        const float* __restrict__ bq, const float* __restrict__ bk,
        const float* __restrict__ bv,
        _Float16* __restrict__ Qh,         // [16384,64] (scaled by 0.125)
        _Float16* __restrict__ Kh,         // [16384,64]
        _Float16* __restrict__ Vt)         // [4][64][4096]
{
    const int wid  = threadIdx.x >> 5;
    const int lane = threadIdx.x & 31;
    const int ml   = lane & 15;
    const int hf   = lane >> 4;
    const int rowbase = (blockIdx.x * 4 + wid) * 16;

    v8f zero = {};
    v8f acc[12];
#pragma unroll
    for (int i = 0; i < 12; ++i) acc[i] = zero;

    const _Float16* xr = Xh + (size_t)(rowbase + ml) * IN_DIM;

    for (int kc = 0; kc < IN_DIM; kc += 32) {
        v16h a = *(const v16h*)(xr + kc + hf * 16);

#pragma unroll
        for (int m = 0; m < 3; ++m) {
            const _Float16* W = Wth + (size_t)m * 32768;
#pragma unroll
            for (int c = 0; c < 4; ++c) {
                v16h b = *(const v16h*)(W + (size_t)(c * 16 + ml) * 512 + hf * 16 + kc);
                acc[m * 4 + c] = __builtin_amdgcn_wmma_f32_16x16x32_f16(
                    false, a, false, b, (short)0, acc[m * 4 + c], false, false);
            }
        }
    }

#pragma unroll
    for (int m = 0; m < 3; ++m) {
        const float* bias = (m == 0) ? bq : (m == 1) ? bk : bv;
        const float scale = (m == 0) ? 0.125f : 1.0f;
#pragma unroll
        for (int c = 0; c < 4; ++c) {
            int d = c * 16 + ml;
            float bb = bias[d];
#pragma unroll
            for (int r = 0; r < 8; ++r) {
                int row = rowbase + r + 8 * hf;
                float v = acc[m * 4 + c][r] + bb;
                if (m == 2) {
                    int b_ = row >> 12, s_ = row & 4095;
                    Vt[((size_t)b_ * 64 + d) * SEQ + s_] = (_Float16)v;
                } else {
                    _Float16* o = (m == 0) ? Qh : Kh;
                    o[(size_t)row * 64 + d] = (_Float16)(v * scale);
                }
            }
        }
    }
}

// ---------------------------------------------------------------------------
// Kernel 2: flash attention. Block = 4 waves, same batch, 4 adjacent q-tiles.
// K/V tiles are staged Global -> LDS with async-to-LDS DMA (ASYNCcnt),
// double-buffered, shared by all 4 waves. Scores/PV via f16 WMMA.
// ---------------------------------------------------------------------------
__global__ void __launch_bounds__(128) k_attn(
        const _Float16* __restrict__ Qh,   // [16384,64] pre-scaled by 1/8
        const _Float16* __restrict__ Kh,   // [16384,64]
        const _Float16* __restrict__ Vt,   // [4][64][4096]
        const int* __restrict__ mask,      // [4096,4096]
        float* __restrict__ out)           // [4,4096,64]
{
    __shared__ __align__(128) _Float16 Klds[2][32][64];   // 8 KB  [buf][key][d]
    __shared__ __align__(128) _Float16 Vlds[2][64][32];   // 8 KB  [buf][d][key]
    __shared__ __align__(32)  _Float16 Plds[4][16][32];   // 4 KB  per-wave P staging

    const int wid  = threadIdx.x >> 5;
    const int lane = threadIdx.x & 31;
    const int ml   = lane & 15;
    const int hf   = lane >> 4;
    const int qtile = blockIdx.x * 4 + wid;
    const int batch = blockIdx.y;
    const int qbase = qtile * 16;
    const size_t rowoff = (size_t)batch * SEQ + qbase;

    const _Float16* Kb = Kh + (size_t)batch * SEQ * 64;
    const _Float16* Vb = Vt + (size_t)batch * 64 * SEQ;

    // ---- per-lane chunk mapping for cooperative async tile copy ----
    // K tile: 32 keys x 64 halves = 256 x 16B chunks; this wave covers 64 of them
    const int idA = wid * 64 + lane;        // chunk ids idA, idA+32
    // precomputed (key, c8) / (d, c4) pairs for the two chunks each of K and V
    const int kKey0 = idA >> 3,        kC80 = idA & 7;
    const int kKey1 = (idA + 32) >> 3, kC81 = (idA + 32) & 7;
    const int vD0 = idA >> 2,          vC40 = idA & 3;
    const int vD1 = (idA + 32) >> 2,   vC41 = (idA + 32) & 3;

    // ---- load Q tile once: two 16x32 A-operand chunks over head-dim 64 ----
    const _Float16* qr = Qh + (rowoff + ml) * 64;
    v16h aq[2];
#pragma unroll
    for (int dc = 0; dc < 2; ++dc) {
        aq[dc] = *(const v16h*)(qr + dc * 32 + hf * 16);
    }

    v8f zero = {};
    v8f O[4];
#pragma unroll
    for (int c = 0; c < 4; ++c) O[c] = zero;
    float runmax[8], runsum[8];
#pragma unroll
    for (int r = 0; r < 8; ++r) { runmax[r] = -3.0e38f; runsum[r] = 0.0f; }

    // async issue of one 32-key K tile + V tile into LDS buffer `buf`
    auto issue_tiles = [&](int buf, int kbn) {
        {   // K chunk 0
            const _Float16* g = Kb + (size_t)(kbn + kKey0) * 64 + kC80 * 8;
            unsigned l = (unsigned)(uintptr_t)&Klds[buf][kKey0][kC80 * 8];
            asm volatile("global_load_async_to_lds_b128 %0, %1, off" :: "v"(l), "v"(g) : "memory");
        }
        {   // K chunk 1
            const _Float16* g = Kb + (size_t)(kbn + kKey1) * 64 + kC81 * 8;
            unsigned l = (unsigned)(uintptr_t)&Klds[buf][kKey1][kC81 * 8];
            asm volatile("global_load_async_to_lds_b128 %0, %1, off" :: "v"(l), "v"(g) : "memory");
        }
        {   // V chunk 0
            const _Float16* g = Vb + (size_t)vD0 * SEQ + kbn + vC40 * 8;
            unsigned l = (unsigned)(uintptr_t)&Vlds[buf][vD0][vC40 * 8];
            asm volatile("global_load_async_to_lds_b128 %0, %1, off" :: "v"(l), "v"(g) : "memory");
        }
        {   // V chunk 1
            const _Float16* g = Vb + (size_t)vD1 * SEQ + kbn + vC41 * 8;
            unsigned l = (unsigned)(uintptr_t)&Vlds[buf][vD1][vC41 * 8];
            asm volatile("global_load_async_to_lds_b128 %0, %1, off" :: "v"(l), "v"(g) : "memory");
        }
    };

    issue_tiles(0, 0);   // prologue: fill buffer 0

    for (int kb = 0; kb < SEQ; kb += 32) {
        const int cur = (kb >> 5) & 1;
        // issue next tile into the other buffer (wraps to 0 on last iter; benign)
        const int kbn = (kb + 32 < SEQ) ? (kb + 32) : 0;
        issue_tiles(cur ^ 1, kbn);

        // async loads complete in-order: <=4 outstanding => current buffer done
        asm volatile("s_wait_asynccnt 0x4" ::: "memory");
        __syncthreads();   // cross-wave visibility of the staged tiles

        // ---- scores for 32 keys from Klds: two 16x16 tiles, 2 WMMAs each ----
        v8f s0 = zero, s1 = zero;
#pragma unroll
        for (int h2 = 0; h2 < 2; ++h2) {
            const _Float16* kp = &Klds[cur][h2 * 16 + ml][hf * 16];
            v16h b0 = *(const v16h*)(kp);        // d 0..31 chunk
            v16h b1 = *(const v16h*)(kp + 32);   // d 32..63 chunk
            if (h2 == 0) {
                s0 = __builtin_amdgcn_wmma_f32_16x16x32_f16(false, aq[0], false, b0, (short)0, s0, false, false);
                s0 = __builtin_amdgcn_wmma_f32_16x16x32_f16(false, aq[1], false, b1, (short)0, s0, false, false);
            } else {
                s1 = __builtin_amdgcn_wmma_f32_16x16x32_f16(false, aq[0], false, b0, (short)0, s1, false, false);
                s1 = __builtin_amdgcn_wmma_f32_16x16x32_f16(false, aq[1], false, b1, (short)0, s1, false, false);
            }
        }

        // ---- mask + online softmax over the 32-key block ----
        float alpha_[8];
#pragma unroll
        for (int r = 0; r < 8; ++r) {
            size_t mbase = (size_t)(qbase + r + 8 * hf) * SEQ + kb;
            int m0 = mask[mbase + ml];
            int m1 = mask[mbase + 16 + ml];
            float v0 = (m0 == 0) ? 1e-6f : s0[r];
            float v1 = (m1 == 0) ? 1e-6f : s1[r];
            float rm = fmaxf(v0, v1);
            rm = fmaxf(rm, __shfl_xor(rm, 1, 32));
            rm = fmaxf(rm, __shfl_xor(rm, 2, 32));
            rm = fmaxf(rm, __shfl_xor(rm, 4, 32));
            rm = fmaxf(rm, __shfl_xor(rm, 8, 32));
            float nm = fmaxf(runmax[r], rm);
            float alpha = __expf(runmax[r] - nm);
            float p0 = __expf(v0 - nm);
            float p1 = __expf(v1 - nm);
            float ps = p0 + p1;
            ps += __shfl_xor(ps, 1, 32);
            ps += __shfl_xor(ps, 2, 32);
            ps += __shfl_xor(ps, 4, 32);
            ps += __shfl_xor(ps, 8, 32);
            runsum[r] = runsum[r] * alpha + ps;
            runmax[r] = nm;
            alpha_[r] = alpha;
            Plds[wid][r + 8 * hf][ml]      = (_Float16)p0;
            Plds[wid][r + 8 * hf][16 + ml] = (_Float16)p1;
        }
#pragma unroll
        for (int c = 0; c < 4; ++c)
#pragma unroll
            for (int r = 0; r < 8; ++r) O[c][r] *= alpha_[r];

        // wait for our own DS stores (CDNA5 split counter) before re-load
        asm volatile("s_wait_dscnt 0" ::: "memory");

        // ---- P (16x32) @ V (32x64) from Vlds: 4 WMMAs ----
        v16h ap;
        {
            const _Float16* pp = &Plds[wid][ml][0];
            ap = *(const v16h*)(pp + hf * 16);
        }
#pragma unroll
        for (int c = 0; c < 4; ++c) {
            const _Float16* vp = &Vlds[cur][c * 16 + ml][hf * 16];
            v16h bvv = *(const v16h*)vp;
            O[c] = __builtin_amdgcn_wmma_f32_16x16x32_f16(false, ap, false, bvv, (short)0, O[c], false, false);
        }

        __syncthreads();   // all waves done reading before buffer is re-filled
    }

    // ---- normalize and store f32 output ----
#pragma unroll
    for (int c = 0; c < 4; ++c) {
        int d = c * 16 + ml;
#pragma unroll
        for (int r = 0; r < 8; ++r) {
            out[(rowoff + r + 8 * hf) * 64 + d] = O[c][r] / runsum[r];
        }
    }
}

// ---------------------------------------------------------------------------
extern "C" void kernel_launch(void* const* d_in, const int* in_sizes, int n_in,
                              void* d_out, int out_size, void* d_ws, size_t ws_size,
                              hipStream_t stream) {
    const float* X  = (const float*)d_in[0];
    const int* mask = (const int*)d_in[1];
    const float* Wq = (const float*)d_in[2];
    const float* bq = (const float*)d_in[3];
    const float* Wk = (const float*)d_in[4];
    const float* bk = (const float*)d_in[5];
    const float* Wv = (const float*)d_in[6];
    const float* bv = (const float*)d_in[7];
    float* out = (float*)d_out;

    char* ws = (char*)d_ws;
    _Float16* Xh  = (_Float16*)(ws);                         // 16 MB
    _Float16* Wth = (_Float16*)(ws + 16777216);              // 192 KB
    _Float16* Qh  = (_Float16*)(ws + 16777216 + 196608);     // 2 MB
    _Float16* Kh  = Qh + (size_t)BATCH * SEQ * OUT_DIM;      // 2 MB
    _Float16* Vt  = Kh + (size_t)BATCH * SEQ * OUT_DIM;      // 2 MB

    k_cvt_x<<<4096, 256, 0, stream>>>(X, Xh, BATCH * SEQ * IN_DIM);
    k_wt<<<(3 * OUT_DIM * IN_DIM + 255) / 256, 256, 0, stream>>>(Wq, Wk, Wv, Wth);
    k_qkv<<<dim3((BATCH * SEQ / 16) / 4), 128, 0, stream>>>(Xh, Wth, bq, bk, bv, Qh, Kh, Vt);
    k_attn<<<dim3((SEQ / 16) / 4, BATCH), 128, 0, stream>>>(Qh, Kh, Vt, mask, out);
}
